// FourierAttention_57810259804525
// MI455X (gfx1250) — matrix operand order repaired and checked by reading, not compiled, not run on hardware
//
#include <hip/hip_runtime.h>
#include <hip/hip_bf16.h>
#include <math.h>

// ---------------- problem constants ----------------
constexpr int CB  = 8;      // batch
constexpr int CH  = 8;      // heads
constexpr int CE  = 64;     // head dim
constexpr int CN  = 1024;   // L = S
constexpr int CF  = 513;    // rfft bins
constexpr int CFP = 544;    // padded bins = 17*32 = 34*16
constexpr int CBH = CB * CH;

typedef __attribute__((ext_vector_type(16))) _Float16 v16h;
typedef __attribute__((ext_vector_type(8)))  _Float16 v8h;
typedef __attribute__((ext_vector_type(2)))  _Float16 v2h;
typedef __attribute__((ext_vector_type(8)))  float    v8f;

union F16x16 { v16h v; v8h h[2]; };

__device__ __forceinline__ v8f wmma16(v16h a, v16h b, v8f c) {
  // D = A(16x32 f16) * B(32x16 f16) + C(16x16 f32)
  return __builtin_amdgcn_wmma_f32_16x16x32_f16(
      /*neg_a=*/false, a, /*neg_b=*/false, b,
      /*c_mod=*/(short)0, c, /*reuse_a=*/false, /*reuse_b=*/false);
}

// A fragment from row-major [m][k] storage (consecutive k), leading dim ld.
// ISA 16-bit A 16x32: lane m = lane&15, hh = lane>>4,
// elems 0..7 -> K = hh*8 + j, elems 8..15 -> K = 16 + hh*8 + (j-8).
__device__ __forceinline__ v16h loadA(const _Float16* p, int ld, int row0, int k0, int lane) {
  int m = lane & 15, hh = lane >> 4;
  const _Float16* r = p + (size_t)(row0 + m) * ld + k0 + hh * 8;
  F16x16 u;
  u.h[0] = *(const v8h*)(r);
  u.h[1] = *(const v8h*)(r + 16);
  return u.v;
}

// B fragment when B is stored transposed [n][k] (consecutive k), leading dim ld.
// ISA 32x16 B: lane n = lane&15, hh = lane>>4,
// elems 0..7 -> K = hh*8 + j, elems 8..15 -> K = 16 + hh*8 + (j-8).
__device__ __forceinline__ v16h loadB_T(const _Float16* p, int ld, int n0, int k0, int lane) {
  int n = lane & 15, hh = lane >> 4;
  const _Float16* r = p + (size_t)(n0 + n) * ld + k0 + hh * 8;
  F16x16 u;
  u.h[0] = *(const v8h*)(r);
  u.h[1] = *(const v8h*)(r + 16);
  return u.v;
}

__device__ __forceinline__ v8f vzero8() {
  v8f z = {0.f, 0.f, 0.f, 0.f, 0.f, 0.f, 0.f, 0.f};
  return z;
}

// ---------------- kernel 0: DFT bases (f16), WMMA-fragment-native layouts ----
// cosF/sinF: [FP][N]  forward rfft basis, stored [f][t]  (B-transposed layout)
// cosI/sinI: [N][FP]  inverse rfft basis, stored [t][f]  (B-transposed layout)
__global__ void k_basis(_Float16* __restrict__ cosF, _Float16* __restrict__ sinF,
                        _Float16* __restrict__ cosI, _Float16* __restrict__ sinI) {
  int idx = blockIdx.x * blockDim.x + threadIdx.x;
  if (idx >= CN * CFP) return;
  int t = idx / CFP, f = idx - t * CFP;
  const float s = 1.0f / 32.0f;  // 1/sqrt(1024), ortho norm
  float c = 0.f, sn = 0.f;
  if (f < CF) {
    int p = (f * t) & (CN - 1);
    float th = 6.28318530717958647692f * (float)p * (1.0f / (float)CN);
    __sincosf(th, &sn, &c);
  }
  bool valid = (f < CF);
  cosF[(size_t)f * CN + t] = (_Float16)(valid ?  c * s : 0.f);
  sinF[(size_t)f * CN + t] = (_Float16)(valid ? -sn * s : 0.f);
  float w = (f == 0 || f == CN / 2) ? 1.f : 2.f;
  cosI[(size_t)t * CFP + f] = (_Float16)(valid ?  w * c * s : 0.f);
  sinI[(size_t)t * CFP + f] = (_Float16)(valid ? -w * sn * s : 0.f);
}

// ---------------- kernel 1: rfft as GEMM ----------------
// x: [B,N,H,E] f32.  out per (b,h): real at +0, imag at +FP*64.
// transpose_out=1 -> store [f][e] (frag-native for scores), 0 -> [e][FP].
// grid (FP/16, BH), block 128 (4 waves, 64x16 tile, K=1024).
__global__ void k_fft(const float* __restrict__ x, _Float16* __restrict__ out,
                      const _Float16* __restrict__ basC, const _Float16* __restrict__ basS,
                      int transpose_out) {
  __shared__ _Float16 As[64 * 32];   // [e][t] tile, f16
  int tid = threadIdx.x, wave = tid >> 5, lane = tid & 31;
  int f0 = blockIdx.x * 16;
  int bh = blockIdx.y, b = bh >> 3, h = bh & 7;
  const float* xb = x + ((size_t)b * CN * CH + h) * CE;  // xb[t*512 + e]
  const size_t PS = (size_t)CFP * 64;
  _Float16* outR = out + (size_t)bh * 2 * PS;
  _Float16* outI = outR + PS;

  v8f accR = vzero8(), accI = vzero8();
  for (int kk = 0; kk < CN; kk += 32) {
    // stage 64(e) x 32(t) input tile -> f16 LDS, packed b32 stores
    for (int i = tid; i < 64 * 16; i += 128) {
      int e = i & 63, tp = i >> 6;             // tp = t/2
      const float* src = xb + (size_t)(kk + 2 * tp) * (CH * CE) + e;
      if (kk + 32 < CN) __builtin_prefetch(src + 32 * (CH * CE), 0, 0);
      v2h pk = {(_Float16)src[0], (_Float16)src[CH * CE]};
      *(v2h*)(&As[e * 32 + 2 * tp]) = pk;
    }
    __syncthreads();
    v16h a  = loadA(As, 32, wave * 16, 0, lane);
    accR = wmma16(a, loadB_T(basC, CN, f0, kk, lane), accR);  // basis [f][t]
    accI = wmma16(a, loadB_T(basS, CN, f0, kk, lane), accI);
    __syncthreads();
  }
  int n = lane & 15, hh = lane >> 4;
  int f = f0 + n;
#pragma unroll
  for (int r = 0; r < 8; ++r) {
    int e = wave * 16 + r + 8 * hh;
    if (transpose_out) {
      outR[(size_t)f * 64 + e] = (_Float16)accR[r];
      outI[(size_t)f * 64 + e] = (_Float16)accI[r];
    } else {
      outR[(size_t)e * CFP + f] = (_Float16)accR[r];
      outI[(size_t)e * CFP + f] = (_Float16)accI[r];
    }
  }
}

// ---------------- kernel 2: scores + |.| + softmax -> attn [x][y] ----------
// QfT/KfT: [BH][2][FP][64] ([f][e]). attn: [BH][FP(x)][FP(y)] row-major.
// grid (FP/16, BH), block 256 (8 waves): 16 x-rows, all 544 y per block.
__global__ void k_scores(const _Float16* __restrict__ QfT, const _Float16* __restrict__ KfT,
                         _Float16* __restrict__ attn) {
  __shared__ float smag[16 * CFP];
  int tid = threadIdx.x, wave = tid >> 5, lane = tid & 31;
  int x0 = blockIdx.x * 16;
  int bh = blockIdx.y;
  const size_t PS = (size_t)CFP * 64;
  const _Float16* Qr = QfT + (size_t)bh * 2 * PS;
  const _Float16* Qi = Qr + PS;
  const _Float16* Kr = KfT + (size_t)bh * 2 * PS;
  const _Float16* Ki = Kr + PS;
  _Float16* At = attn + (size_t)bh * CFP * CFP;

  v16h qr0 = loadA(Qr, 64, x0, 0, lane), qr1 = loadA(Qr, 64, x0, 32, lane);
  v16h qi0 = loadA(Qi, 64, x0, 0, lane), qi1 = loadA(Qi, 64, x0, 32, lane);
  v16h qn0 = -qr0, qn1 = -qr1;  // pre-negated for the imag chain

  for (int yt = wave; yt < CFP / 16; yt += 8) {
    int y0 = yt * 16;
    v16h kr0 = loadB_T(Kr, 64, y0, 0, lane), kr1 = loadB_T(Kr, 64, y0, 32, lane);
    v16h ki0 = loadB_T(Ki, 64, y0, 0, lane), ki1 = loadB_T(Ki, 64, y0, 32, lane);
    // scores = qf * conj(kf): re = qr.kr + qi.ki ; im = qi.kr - qr.ki (sum over e)
    v8f re = wmma16(qr0, kr0, wmma16(qr1, kr1, wmma16(qi0, ki0, wmma16(qi1, ki1, vzero8()))));
    v8f im = wmma16(qi0, kr0, wmma16(qi1, kr1, wmma16(qn0, ki0, wmma16(qn1, ki1, vzero8()))));
    int n = lane & 15, hh = lane >> 4;
    int y = y0 + n;
#pragma unroll
    for (int r = 0; r < 8; ++r) {
      float mag = sqrtf(re[r] * re[r] + im[r] * im[r]) * 0.125f;  // /sqrt(E), T=1
      if (y >= CF) mag = -1e30f;                                  // mask padded keys
      smag[(r + 8 * hh) * CFP + y] = mag;
    }
  }
  __syncthreads();
  // row softmax over 544 (masked) columns; 16 threads per row
  int row = tid >> 4, l16 = tid & 15;
  float* rowp = &smag[row * CFP];
  float mx = -1e30f;
  for (int j = l16; j < CFP; j += 16) mx = fmaxf(mx, rowp[j]);
#pragma unroll
  for (int off = 8; off; off >>= 1) mx = fmaxf(mx, __shfl_xor(mx, off, 16));
  float sum = 0.f;
  for (int j = l16; j < CFP; j += 16) sum += __expf(rowp[j] - mx);
#pragma unroll
  for (int off = 8; off; off >>= 1) sum += __shfl_xor(sum, off, 16);
  float inv = 1.0f / sum;
  // untransposed [x][y] write: coalesced, and row-major [x][y] == B-transposed
  // layout for the attn x Vf GEMM (n=x, k=y).
  for (int j = l16; j < CFP; j += 16)
    At[(size_t)(x0 + row) * CFP + j] = (_Float16)(__expf(rowp[j] - mx) * inv);
}

// ---------------- kernel 3: outf = attn x Vf (LDS-free) ----------------
// Vf/Of: [BH][2][64][FP] ([e][f]); attn [x][y]. Or[e,x] = sum_y Vr[e,y]*attn[x,y].
// grid (FP/16, BH), block 128.
__global__ void k_outf(const _Float16* __restrict__ Vf, const _Float16* __restrict__ attn,
                       _Float16* __restrict__ Of) {
  int tid = threadIdx.x, wave = tid >> 5, lane = tid & 31;
  int x0 = blockIdx.x * 16;
  int bh = blockIdx.y;
  const size_t PS = (size_t)64 * CFP;
  const _Float16* Vr = Vf + (size_t)bh * 2 * PS;
  const _Float16* Vi = Vr + PS;
  const _Float16* At = attn + (size_t)bh * CFP * CFP;
  _Float16* Or = Of + (size_t)bh * 2 * PS;
  _Float16* Oi = Or + PS;

  v8f accR = vzero8(), accI = vzero8();
  for (int kk = 0; kk < CFP; kk += 32) {
    v16h b = loadB_T(At, CFP, x0, kk, lane);       // B[k=y][n=x] = attn[x][y]
    accR = wmma16(loadA(Vr, CFP, wave * 16, kk, lane), b, accR);
    accI = wmma16(loadA(Vi, CFP, wave * 16, kk, lane), b, accI);
  }
  int n = lane & 15, hh = lane >> 4;
#pragma unroll
  for (int r = 0; r < 8; ++r) {
    int e = wave * 16 + r + 8 * hh;
    Or[(size_t)e * CFP + x0 + n] = (_Float16)accR[r];
    Oi[(size_t)e * CFP + x0 + n] = (_Float16)accI[r];
  }
}

// ---------------- kernel 4: irfft as GEMM + transpose (LDS-free) ----------
// out[e,t] = sum_f Or[e,f]*cosI[t][f] + Oi[e,f]*sinI[t][f].
// grid (N/16, BH), block 128.
__global__ void k_irfft(const _Float16* __restrict__ Of,
                        const _Float16* __restrict__ cosI, const _Float16* __restrict__ sinI,
                        float* __restrict__ out) {
  int tid = threadIdx.x, wave = tid >> 5, lane = tid & 31;
  int t0 = blockIdx.x * 16;
  int bh = blockIdx.y, b = bh >> 3, h = bh & 7;
  const size_t PS = (size_t)64 * CFP;
  const _Float16* Or = Of + (size_t)bh * 2 * PS;
  const _Float16* Oi = Or + PS;

  v8f acc = vzero8();
  for (int kk = 0; kk < CFP; kk += 32) {
    acc = wmma16(loadA(Or, CFP, wave * 16, kk, lane), loadB_T(cosI, CFP, t0, kk, lane), acc);
    acc = wmma16(loadA(Oi, CFP, wave * 16, kk, lane), loadB_T(sinI, CFP, t0, kk, lane), acc);
  }
  int n = lane & 15, hh = lane >> 4;
  int t = t0 + n;
#pragma unroll
  for (int r = 0; r < 8; ++r) {
    int e = wave * 16 + r + 8 * hh;
    out[(((size_t)b * CN + t) * CH + h) * CE + e] = acc[r];
  }
}

// ---------------- host launch ----------------
extern "C" void kernel_launch(void* const* d_in, const int* in_sizes, int n_in,
                              void* d_out, int out_size, void* d_ws, size_t ws_size,
                              hipStream_t stream) {
  const float* q = (const float*)d_in[0];
  const float* k = (const float*)d_in[1];
  const float* v = (const float*)d_in[2];
  float* out = (float*)d_out;

  char* ws = (char*)d_ws;
  auto take = [&](size_t bytes) -> void* {
    void* p = ws;
    ws += (bytes + 255) & ~(size_t)255;
    return p;
  };
  const size_t PS = (size_t)CFP * 64;                       // per (bh, part) matrix
  _Float16* cosF = (_Float16*)take((size_t)CFP * CN * 2);       // [f][t]
  _Float16* sinF = (_Float16*)take((size_t)CFP * CN * 2);       // [f][t]
  _Float16* cosI = (_Float16*)take((size_t)CN * CFP * 2);       // [t][f]
  _Float16* sinI = (_Float16*)take((size_t)CN * CFP * 2);       // [t][f]
  _Float16* QfT  = (_Float16*)take((size_t)CBH * 2 * PS * 2);   // [bh][ri][f][e]
  _Float16* KfT  = (_Float16*)take((size_t)CBH * 2 * PS * 2);   // [bh][ri][f][e]
  _Float16* Vf   = (_Float16*)take((size_t)CBH * 2 * PS * 2);   // [bh][ri][e][f]
  _Float16* Of   = (_Float16*)take((size_t)CBH * 2 * PS * 2);   // [bh][ri][e][f]
  _Float16* attn = (_Float16*)take((size_t)CBH * CFP * CFP * 2);// [bh][x][y]

  {
    int total = CN * CFP;
    k_basis<<<(total + 255) / 256, 256, 0, stream>>>(cosF, sinF, cosI, sinI);
  }
  dim3 gF(CFP / 16, CBH);
  k_fft<<<gF, 128, 0, stream>>>(q, QfT, cosF, sinF, 1);
  k_fft<<<gF, 128, 0, stream>>>(k, KfT, cosF, sinF, 1);
  k_fft<<<gF, 128, 0, stream>>>(v, Vf,  cosF, sinF, 0);
  k_scores<<<gF, 256, 0, stream>>>(QfT, KfT, attn);
  k_outf<<<gF, 128, 0, stream>>>(Vf, attn, Of);
  dim3 gI(CN / 16, CBH);
  k_irfft<<<gI, 128, 0, stream>>>(Of, cosI, sinI, out);
}
